// ASTNN_16003048145659
// MI455X (gfx1250) — matrix operand order repaired and checked by reading, not compiled, not run on hardware
//
#include <hip/hip_runtime.h>
#include <hip/hip_bf16.h>
#include <math.h>

// ---------------- CDNA5 WMMA plumbing ----------------
typedef __attribute__((ext_vector_type(16))) __bf16 v16bf;
typedef __attribute__((ext_vector_type(8)))  float  v8f;

union Frag { v16bf v; unsigned u[8]; };

__device__ __forceinline__ unsigned bf16b(float f) {
    unsigned u = __float_as_uint(f);
    return (u + 0x7FFFu + ((u >> 16) & 1u)) >> 16;   // round-to-nearest-even
}
__device__ __forceinline__ unsigned pk2(float lo, float hi) {
    return bf16b(lo) | (bf16b(hi) << 16);
}

// Problem constants
#define BATCH 128
#define SEQ   200
#define TREES (BATCH*SEQ)      // 25600
#define KNODE 32
#define EDIM  128
#define ENC   128
#define HID   100
#define G3H   300              // 3*HID
#define OUTD  1000
#define PPAD  224              // 2*HID padded to 7 K-chunks of 32

// -------------------------------------------------------------------------
// Kernel 1: embedding gather + Wc projection (WMMA bf16) + tree accumulate
//           + per-tree max + relu.  4 waves/block, 1 tree per wave.
// -------------------------------------------------------------------------
__global__ __launch_bounds__(128) void k_encode(const int* __restrict__ tokens,
        const float* __restrict__ emb, const float* __restrict__ Wc,
        const float* __restrict__ bc, unsigned short* __restrict__ xbf)
{
    __shared__ __align__(16) unsigned short wclds[ENC*EDIM];   // Wc as bf16 [n][k]
    __shared__ __align__(16) float hcol[4*32*17];              // per-wave C staging
    __shared__ int toklds[128];

    const int tid = threadIdx.x;
    for (int i = tid; i < ENC*EDIM; i += 128) wclds[i] = (unsigned short)bf16b(Wc[i]);
    toklds[tid] = tokens[(long)blockIdx.x*128 + tid];
    __syncthreads();

    const int w = tid >> 5, l = tid & 31;
    const int lh = (l >> 4) & 1, ln = l & 15;
    const long t = (long)blockIdx.x*4 + w;

    // A fragments: 32 node-embedding rows, converted to bf16 on the fly
    Frag a[2][4];
#pragma unroll
    for (int mt = 0; mt < 2; ++mt) {
        const int r = mt*16 + ln;
        const float* row = emb + (long)toklds[w*32 + r] * EDIM;
#pragma unroll
        for (int kt = 0; kt < 4; ++kt)
#pragma unroll
            for (int v = 0; v < 8; ++v) {
                const int k = kt*32 + ((v < 4) ? 2*v : 2*v + 8) + (lh << 3);
                const float2 p = *(const float2*)(row + k);
                a[mt][kt].u[v] = pk2(p.x, p.y);
            }
    }

    float* hc = hcol + w*32*17;
    for (int nt = 0; nt < 8; ++nt) {
        const int col = nt*16 + ln;
        const float cb = bc[col];
        v8f acc0 = {cb,cb,cb,cb,cb,cb,cb,cb};
        v8f acc1 = acc0;
#pragma unroll
        for (int kt = 0; kt < 4; ++kt) {
            Frag b;
#pragma unroll
            for (int v = 0; v < 8; ++v) {
                const int k = kt*32 + ((v < 4) ? 2*v : 2*v + 8) + (lh << 3);
                b.u[v] = *(const unsigned*)(wclds + col*EDIM + k);
            }
            acc0 = __builtin_amdgcn_wmma_f32_16x16x32_bf16(false, a[0][kt].v, false, b.v, (short)0, acc0, false, false);
            acc1 = __builtin_amdgcn_wmma_f32_16x16x32_bf16(false, a[1][kt].v, false, b.v, (short)0, acc1, false, false);
        }
#pragma unroll
        for (int g = 0; g < 8; ++g) {
            hc[(g + 8*lh)*17 + ln]      = acc0[g];
            hc[(16 + g + 8*lh)*17 + ln] = acc1[g];
        }
        __syncthreads();
        if (l < 16) {   // lane == output channel within the tile
            float hv[32];
#pragma unroll
            for (int j = 0; j < 32; ++j) hv[j] = hc[j*17 + l];
#pragma unroll
            for (int j = 31; j >= 1; --j) hv[(j-1) >> 2] += hv[j];   // deepest-first accumulate
            float m = hv[0];
#pragma unroll
            for (int j = 1; j < 32; ++j) m = fmaxf(m, hv[j]);
            m = fmaxf(m, 0.0f);
            xbf[t*ENC + nt*16 + l] = (unsigned short)bf16b(m);
        }
        __syncthreads();
    }
}

// -------------------------------------------------------------------------
// Kernel 2: input-gate projections for both GRU directions (no seq dep):
//           G = X @ Wih^T + bih.  M=25600, K=128, N=300 (19 tiles).
// -------------------------------------------------------------------------
__global__ __launch_bounds__(128) void k_gin(const unsigned short* __restrict__ xbf,
        const float* __restrict__ Wf, const float* __restrict__ bf,
        const float* __restrict__ Wb, const float* __restrict__ bb,
        float* __restrict__ Gf, float* __restrict__ Gb)
{
    const int dir = blockIdx.y;
    const float* W  = dir ? Wb : Wf;
    const float* bi = dir ? bb : bf;
    float* G = dir ? Gb : Gf;

    const int tid = threadIdx.x, w = tid >> 5, l = tid & 31;
    const int lh = (l >> 4) & 1, ln = l & 15;
    const int rbase = blockIdx.x*64 + w*16;

    Frag a[4];
    {
        const unsigned short* row = xbf + (long)(rbase + ln)*ENC;
#pragma unroll
        for (int kt = 0; kt < 4; ++kt)
#pragma unroll
            for (int v = 0; v < 8; ++v) {
                const int k = kt*32 + ((v < 4) ? 2*v : 2*v + 8) + (lh << 3);
                a[kt].u[v] = *(const unsigned*)(row + k);
            }
    }
    for (int nt = 0; nt < 19; ++nt) {
        const int col  = nt*16 + ln;
        const int colc = col < G3H ? col : G3H - 1;
        const float cb = col < G3H ? bi[col] : 0.0f;
        v8f acc = {cb,cb,cb,cb,cb,cb,cb,cb};
#pragma unroll
        for (int kt = 0; kt < 4; ++kt) {
            Frag b;
#pragma unroll
            for (int v = 0; v < 8; ++v) {
                const int k = kt*32 + ((v < 4) ? 2*v : 2*v + 8) + (lh << 3);
                const float2 p = *(const float2*)(W + (long)colc*ENC + k);
                b.u[v] = pk2(p.x, p.y);
            }
            acc = __builtin_amdgcn_wmma_f32_16x16x32_bf16(false, a[kt].v, false, b.v, (short)0, acc, false, false);
        }
        if (col < G3H)
#pragma unroll
            for (int g = 0; g < 8; ++g)
                G[(long)(rbase + g + 8*lh)*G3H + col] = acc[g];
    }
}

// -------------------------------------------------------------------------
// Kernel 3: sequential GRU scan.  grid = (8 batch tiles, 2 directions),
//           one wave each.  Per step: gh = h @ Whh^T + bhh via WMMA
//           (K=100 zero-padded to 128), then elementwise GRU update.
// -------------------------------------------------------------------------
__global__ __launch_bounds__(32) void k_gru(const float* __restrict__ Gf,
        const float* __restrict__ Gb,
        const float* __restrict__ Whf, const float* __restrict__ bhf,
        const float* __restrict__ Whb, const float* __restrict__ bhb,
        float* __restrict__ outf, float* __restrict__ outb)
{
    __shared__ __align__(16) float hlds[16*HID];
    __shared__ __align__(16) float gh[16*304];

    const int dir = blockIdx.y;
    const float* G  = dir ? Gb  : Gf;
    const float* Wh = dir ? Whb : Whf;
    const float* bh = dir ? bhb : bhf;
    float* Out = dir ? outb : outf;

    const int l  = threadIdx.x;
    const int lh = (l >> 4) & 1, ln = l & 15;
    const int bbase = blockIdx.x*16;

    for (int i = l; i < 16*HID; i += 32) hlds[i] = 0.0f;
    __syncthreads();

    for (int step = 0; step < SEQ; ++step) {
        const int s = dir ? (SEQ - 1 - step) : step;

        Frag a[4];  // h state as bf16, K padded 100 -> 128
#pragma unroll
        for (int kt = 0; kt < 4; ++kt)
#pragma unroll
            for (int v = 0; v < 8; ++v) {
                const int k = kt*32 + ((v < 4) ? 2*v : 2*v + 8) + (lh << 3);
                if (k < HID) {
                    const float2 p = *(const float2*)(hlds + ln*HID + k);
                    a[kt].u[v] = pk2(p.x, p.y);
                } else a[kt].u[v] = 0u;
            }

        for (int nt = 0; nt < 19; ++nt) {
            const int col  = nt*16 + ln;
            const int colc = col < G3H ? col : G3H - 1;
            const float cb = col < G3H ? bh[col] : 0.0f;
            v8f acc = {cb,cb,cb,cb,cb,cb,cb,cb};
#pragma unroll
            for (int kt = 0; kt < 4; ++kt) {
                Frag b;
#pragma unroll
                for (int v = 0; v < 8; ++v) {
                    const int k = kt*32 + ((v < 4) ? 2*v : 2*v + 8) + (lh << 3);
                    if (k < HID) {
                        const float2 p = *(const float2*)(Wh + (long)colc*HID + k);
                        b.u[v] = pk2(p.x, p.y);
                    } else b.u[v] = 0u;
                }
                acc = __builtin_amdgcn_wmma_f32_16x16x32_bf16(false, a[kt].v, false, b.v, (short)0, acc, false, false);
            }
#pragma unroll
            for (int g = 0; g < 8; ++g)
                gh[(g + 8*lh)*304 + col] = acc[g];
        }
        __syncthreads();

        for (int idx = l; idx < 16*HID; idx += 32) {
            const int m = idx / HID, j = idx % HID;
            const long grow = ((long)(bbase + m)*SEQ + s)*G3H;
            const float ir  = G[grow + j];
            const float iz  = G[grow + HID + j];
            const float inn = G[grow + 2*HID + j];
            const float r = 1.0f / (1.0f + expf(-(ir + gh[m*304 + j])));
            const float z = 1.0f / (1.0f + expf(-(iz + gh[m*304 + HID + j])));
            const float n = tanhf(inn + r * gh[m*304 + 2*HID + j]);
            const float hprev = hlds[m*HID + j];
            const float hnew  = (1.0f - z)*n + z*hprev;
            hlds[m*HID + j] = hnew;
            Out[((long)(bbase + m)*SEQ + s)*HID + j] = hnew;
        }
        if (step + 1 < SEQ) {   // warm L2 for next step's gate rows
            const int s2 = dir ? (SEQ - 2 - step) : step + 1;
            __builtin_prefetch(G + ((long)(bbase + ln)*SEQ + s2)*G3H, 0, 1);
        }
        __syncthreads();
    }
}

// -------------------------------------------------------------------------
// Kernel 4: max-pool over time -> bf16, K padded to 224 with zeros.
// -------------------------------------------------------------------------
__global__ __launch_bounds__(256) void k_pool(const float* __restrict__ outf,
        const float* __restrict__ outb, unsigned short* __restrict__ pbf)
{
    const int id = blockIdx.x*256 + threadIdx.x;
    if (id >= BATCH*PPAD) return;
    const int b = id / PPAD, c = id % PPAD;
    float m = 0.0f;
    if (c < 2*HID) {
        const float* src = (c < HID) ? outf : outb;
        const int j = (c < HID) ? c : c - HID;
        m = -3.402823466e38f;
        for (int s = 0; s < SEQ; ++s)
            m = fmaxf(m, src[((long)b*SEQ + s)*HID + j]);
    }
    pbf[(long)b*PPAD + c] = (unsigned short)bf16b(m);
}

// -------------------------------------------------------------------------
// Kernel 5: out = pooled @ Wout^T + bout  (M=128, K=200->224, N=1000)
// -------------------------------------------------------------------------
__global__ __launch_bounds__(128) void k_out(const unsigned short* __restrict__ pbf,
        const float* __restrict__ Wout, const float* __restrict__ bout,
        float* __restrict__ out)
{
    const int tid = threadIdx.x, w = tid >> 5, l = tid & 31;
    const int lh = (l >> 4) & 1, ln = l & 15;
    const int rbase = blockIdx.x*64 + w*16;

    Frag a[7];
    {
        const unsigned short* row = pbf + (long)(rbase + ln)*PPAD;
#pragma unroll
        for (int kt = 0; kt < 7; ++kt)
#pragma unroll
            for (int v = 0; v < 8; ++v) {
                const int k = kt*32 + ((v < 4) ? 2*v : 2*v + 8) + (lh << 3);
                a[kt].u[v] = *(const unsigned*)(row + k);
            }
    }
    for (int nt = 0; nt < 63; ++nt) {
        const int col  = nt*16 + ln;
        const int colc = col < OUTD ? col : OUTD - 1;
        const float cb = col < OUTD ? bout[col] : 0.0f;
        v8f acc = {cb,cb,cb,cb,cb,cb,cb,cb};
#pragma unroll
        for (int kt = 0; kt < 7; ++kt) {
            Frag b;
#pragma unroll
            for (int v = 0; v < 8; ++v) {
                const int k = kt*32 + ((v < 4) ? 2*v : 2*v + 8) + (lh << 3);
                if (k < 2*HID) {
                    const float2 p = *(const float2*)(Wout + (long)colc*(2*HID) + k);
                    b.u[v] = pk2(p.x, p.y);
                } else b.u[v] = 0u;
            }
            acc = __builtin_amdgcn_wmma_f32_16x16x32_bf16(false, a[kt].v, false, b.v, (short)0, acc, false, false);
        }
        if (col < OUTD)
#pragma unroll
            for (int g = 0; g < 8; ++g)
                out[(long)(rbase + g + 8*lh)*OUTD + col] = acc[g];
    }
}

// -------------------------------------------------------------------------
extern "C" void kernel_launch(void* const* d_in, const int* in_sizes, int n_in,
                              void* d_out, int out_size, void* d_ws, size_t ws_size,
                              hipStream_t stream) {
    const int*   tokens = (const int*)  d_in[0];
    const float* emb    = (const float*)d_in[4];
    const float* Wc     = (const float*)d_in[5];
    const float* bc     = (const float*)d_in[6];
    const float* Wih_f  = (const float*)d_in[7];
    const float* Whh_f  = (const float*)d_in[8];
    const float* bih_f  = (const float*)d_in[9];
    const float* bhh_f  = (const float*)d_in[10];
    const float* Wih_b  = (const float*)d_in[11];
    const float* Whh_b  = (const float*)d_in[12];
    const float* bih_b  = (const float*)d_in[13];
    const float* bhh_b  = (const float*)d_in[14];
    const float* Wout   = (const float*)d_in[15];
    const float* bout   = (const float*)d_in[16];

    // Workspace layout (bytes)
    const size_t XBF_B = (size_t)TREES*ENC*2;        //  6,553,600
    const size_t G_B   = (size_t)TREES*G3H*4;        // 30,720,000
    const size_t OUT_B = (size_t)BATCH*SEQ*HID*4;    // 10,240,000
    const size_t PBF_B = (size_t)BATCH*PPAD*2;       //     57,344
    if (ws_size < XBF_B + 2*G_B + 2*OUT_B + PBF_B) return;

    char* ws = (char*)d_ws;
    unsigned short* xbf  = (unsigned short*)ws;
    float* Gf   = (float*)(ws + XBF_B);
    float* Gb   = (float*)(ws + XBF_B + G_B);
    float* outf = (float*)(ws + XBF_B + 2*G_B);
    float* outb = (float*)(ws + XBF_B + 2*G_B + OUT_B);
    unsigned short* pbf = (unsigned short*)(ws + XBF_B + 2*G_B + 2*OUT_B);

    k_encode<<<TREES/4, 128, 0, stream>>>(tokens, emb, Wc, bc, xbf);
    k_gin<<<dim3(TREES/64, 2), 128, 0, stream>>>(xbf, Wih_f, bih_f, Wih_b, bih_b, Gf, Gb);
    k_gru<<<dim3(BATCH/16, 2), 32, 0, stream>>>(Gf, Gb, Whh_f, bhh_f, Whh_b, bhh_b, outf, outb);
    k_pool<<<(BATCH*PPAD + 255)/256, 256, 0, stream>>>(outf, outb, pbf);
    k_out<<<2, 128, 0, stream>>>(pbf, Wout, bout, (float*)d_out);
}
